// MultiTrans_83820581749073
// MI455X (gfx1250) — matrix operand order repaired and checked by reading, not compiled, not run on hardware
//
#include <hip/hip_runtime.h>

// MI455X / gfx1250, wave32. All large GEMMs via v_wmma_f32_16x16x32_f16.
// LDS tiles stored in WMMA fragment order (2x ds_load_b128 per operand),
// double-buffered ping-pong staging (1 barrier/iteration), 2 WMMAs per wave
// per iteration (A fragment reused over two B tiles).

typedef __attribute__((ext_vector_type(16))) _Float16 v16h;
typedef __attribute__((ext_vector_type(8)))  float    v8f;
typedef __attribute__((ext_vector_type(4)))  unsigned v4u;

constexpr int nB = 32, nC = 512, nH = 16, nW = 16, nN = 256;
constexpr size_t BCN = (size_t)nB * nC * nN;               // 4,194,304 elems

// Fragment layout (CDNA5 ISA):
//  A 16x32 (MxK): lane = m + 16*((k>>3)&1), elem e = (k>>4)*8 + (k&7)
//  B 32x16 (KxN): lane = n + 16*((k>>3)&1), elem e = (k>>4)*8 + (k&7)
// Tiles stored as [lane][16] f16 -> 32B contiguous per lane.

static __device__ __forceinline__ v16h ld_frag(const _Float16* p) {
  union { v4u u[2]; v16h h; } r;
  const v4u* q = (const v4u*)p;
  r.u[0] = q[0];
  r.u[1] = q[1];
  return r.h;
}
static __device__ __forceinline__ void st_frag(_Float16* p, v16h f) {
  union { v4u u[2]; v16h h; } r;
  r.h = f;
  v4u* q = (v4u*)p;
  q[0] = r.u[0];
  q[1] = r.u[1];
}
static __device__ __forceinline__ void st_pair(_Float16* p, _Float16 a, _Float16 b) {
  union { _Float16 h[2]; unsigned u; } pk;
  pk.h[0] = a; pk.h[1] = b;
  *(unsigned*)p = pk.u;
}
static __device__ __forceinline__ v8f wmma_f16(v16h a, v16h b, v8f c) {
  return __builtin_amdgcn_wmma_f32_16x16x32_f16(false, a, false, b, (short)0, c, false, false);
}

// ---------- Kernel 1: QKV projection  Y[o,n] = sum_c W[o,c] X[c,n] + bias (+pos for K) ----------
// grid (8 n-pairs, 4 o-tiles, 32 b), block 256 (8 waves -> 128x32 tile)
__global__ __launch_bounds__(256) void qkv_kernel(
    const float* __restrict__ X, const float* __restrict__ Wt,
    const float* __restrict__ bias, const float* __restrict__ rH,
    const float* __restrict__ rW, int addPos, _Float16* __restrict__ out) {
  __shared__ __align__(16) _Float16 sA[2][8 * 512];
  __shared__ __align__(16) _Float16 sB[2][2 * 512];
  const int tile_n = blockIdx.x * 32;
  const int o_base = blockIdx.y * 128;
  const int b = blockIdx.z;
  const int t = threadIdx.x;
  const int wv = t >> 5, lane = t & 31;
  const float* Xb = X + (size_t)b * nC * nN;

  auto stage = [&](int kc, _Float16* dA, _Float16* dB) {
    {
      const int s = t >> 5, l = t & 31;
      const int m = l & 15, kh8 = (l >> 4) * 8;
      const float* rowp = Wt + (size_t)(o_base + s * 16 + m) * nC + kc;
      v16h f;
#pragma unroll
      for (int i = 0; i < 8; ++i) f[i] = (_Float16)rowp[kh8 + i];
#pragma unroll
      for (int i = 0; i < 8; ++i) f[8 + i] = (_Float16)rowp[16 + kh8 + i];
      st_frag(dA + s * 512 + l * 16, f);
    }
    {
      const int l = t & 31, eh = t >> 5;
      const int n0 = l & 15, k = (l >> 4) * 8 + eh * 2 + (eh >> 2) * 8;
#pragma unroll
      for (int nt = 0; nt < 2; ++nt)
        st_pair(dB + nt * 512 + l * 16 + eh * 2,
                (_Float16)Xb[(size_t)(kc + k) * nN + tile_n + nt * 16 + n0],
                (_Float16)Xb[(size_t)(kc + k + 1) * nN + tile_n + nt * 16 + n0]);
    }
  };

  stage(0, sA[0], sB[0]);
  __syncthreads();
  v8f acc0 = {}, acc1 = {};
  int cur = 0;
  for (int kc = 0; kc < nC; kc += 32) {
    if (kc + 32 < nC) stage(kc + 32, sA[cur ^ 1], sB[cur ^ 1]);
    v16h a = ld_frag(sA[cur] + wv * 512 + lane * 16);
    acc0 = wmma_f16(a, ld_frag(sB[cur] + lane * 16), acc0);
    acc1 = wmma_f16(a, ld_frag(sB[cur] + 512 + lane * 16), acc1);
    __syncthreads();
    cur ^= 1;
  }
  const int half = lane >> 4, nl = lane & 15;
  v8f accs[2] = {acc0, acc1};
#pragma unroll
  for (int nt = 0; nt < 2; ++nt) {
    const int n = tile_n + nt * 16 + nl;
#pragma unroll
    for (int r = 0; r < 8; ++r) {
      const int o = o_base + wv * 16 + r + 8 * half;
      float vv = accs[nt][r] + bias[o];
      if (addPos) vv += rH[o * nH + (n >> 4)] + rW[o * nW + (n & 15)];
      out[((size_t)b * nC + o) * nN + n] = (_Float16)vv;
    }
  }
}

// ---------- Kernel 2: energy = Q^T K', row-softmax -> attn (f16) ----------
// grid (16 n-tiles, 32 b), block 512 (16 waves, one 16(m)-tile each)
__global__ __launch_bounds__(512) void attn_kernel(const _Float16* __restrict__ Q,
                                                   const _Float16* __restrict__ K,
                                                   _Float16* __restrict__ attn) {
  __shared__ __align__(16) _Float16 sQ[2][512];
  __shared__ __align__(16) _Float16 sK[2][16 * 512];
  __shared__ float sE[16 * 256];
  const int n_base = blockIdx.x * 16;
  const int b = blockIdx.y;
  const int t = threadIdx.x;
  const int wv = t >> 5, lane = t & 31;
  const _Float16* Qb = Q + (size_t)b * nC * nN;
  const _Float16* Kb = K + (size_t)b * nC * nN;

  auto stage = [&](int kc, _Float16* dQ, _Float16* dK) {
    {
      const int l = t & 31, e = t >> 5;                 // e 0..15
      const int m = l & 15, kh8 = (l >> 4) * 8;
      const int k = (l >> 4) * 16 + e;
      dQ[l * 16 + e] = Qb[(size_t)(kc + k) * nN + n_base + m];
    }
    {
      const int tile = t >> 5, l = t & 31;
      const int n0 = l & 15, kh = (l >> 4) * 16;
      v16h f;
#pragma unroll
      for (int e = 0; e < 16; ++e)
        f[e] = Kb[(size_t)(kc + kh + e) * nN + tile * 16 + n0];
      st_frag(dK + tile * 512 + l * 16, f);
    }
  };

  stage(0, sQ[0], sK[0]);
  __syncthreads();
  v8f acc = {};
  int cur = 0;
  for (int kc = 0; kc < nC; kc += 32) {
    if (kc + 32 < nC) stage(kc + 32, sQ[cur ^ 1], sK[cur ^ 1]);
    acc = wmma_f16(ld_frag(sQ[cur] + lane * 16),
                   ld_frag(sK[cur] + wv * 512 + lane * 16), acc);
    __syncthreads();
    cur ^= 1;
  }
  const int half = lane >> 4, ml = lane & 15;
#pragma unroll
  for (int r = 0; r < 8; ++r) sE[(r + 8 * half) * 256 + wv * 16 + ml] = acc[r];
  __syncthreads();
  // softmax: wave wv owns row wv (wave32 shuffle reduction)
  {
    const int row = wv;
    float vals[8];
    float mx = -1e30f;
#pragma unroll
    for (int i = 0; i < 8; ++i) {
      vals[i] = sE[row * 256 + lane + i * 32];
      mx = fmaxf(mx, vals[i]);
    }
#pragma unroll
    for (int off = 16; off >= 1; off >>= 1) mx = fmaxf(mx, __shfl_xor(mx, off, 32));
    float sum = 0.f;
#pragma unroll
    for (int i = 0; i < 8; ++i) { vals[i] = __expf(vals[i] - mx); sum += vals[i]; }
#pragma unroll
    for (int off = 16; off >= 1; off >>= 1) sum += __shfl_xor(sum, off, 32);
    const float inv = 1.0f / sum;
    const int n = n_base + row;
#pragma unroll
    for (int i = 0; i < 8; ++i)
      attn[((size_t)b * nN + n) * nN + lane + i * 32] = (_Float16)(vals[i] * inv);
  }
}

// ---------- Kernel 3: out[c,n] = sum_m V[c,m] attn[n,m] ----------
// grid (8 n-pairs, 4 c-tiles, 32 b), block 256 (128c x 32n per workgroup)
__global__ __launch_bounds__(256) void av_kernel(const _Float16* __restrict__ V,
                                                 const _Float16* __restrict__ attn,
                                                 _Float16* __restrict__ out) {
  __shared__ __align__(16) _Float16 sV[2][8 * 512];
  __shared__ __align__(16) _Float16 sT[2][2 * 512];
  const int n_base = blockIdx.x * 32;
  const int c_base = blockIdx.y * 128;
  const int b = blockIdx.z;
  const int t = threadIdx.x;
  const int wv = t >> 5, lane = t & 31;

  auto stage = [&](int km, _Float16* dV, _Float16* dT) {
    {
      const int s = t >> 5, l = t & 31;
      const int m = l & 15, kh8 = (l >> 4) * 8;
      const _Float16* rowp = V + ((size_t)b * nC + c_base + s * 16 + m) * nN + km;
      v16h f;
#pragma unroll
      for (int i = 0; i < 8; ++i) f[i] = rowp[kh8 + i];
#pragma unroll
      for (int i = 0; i < 8; ++i) f[8 + i] = rowp[16 + kh8 + i];
      st_frag(dV + s * 512 + l * 16, f);
    }
    {
      const int l = t & 31, eh = t >> 5;
      const int n0 = l & 15, k = (l >> 4) * 8 + eh * 2 + (eh >> 2) * 8;
#pragma unroll
      for (int nt = 0; nt < 2; ++nt) {
        const _Float16* src =
            attn + ((size_t)b * nN + n_base + nt * 16 + n0) * nN + km + k;
        st_pair(dT + nt * 512 + l * 16 + eh * 2, src[0], src[1]);
      }
    }
  };

  stage(0, sV[0], sT[0]);
  __syncthreads();
  v8f acc0 = {}, acc1 = {};
  int cur = 0;
  for (int km = 0; km < nN; km += 32) {
    if (km + 32 < nN) stage(km + 32, sV[cur ^ 1], sT[cur ^ 1]);
    v16h a = ld_frag(sV[cur] + wv * 512 + lane * 16);
    acc0 = wmma_f16(a, ld_frag(sT[cur] + lane * 16), acc0);
    acc1 = wmma_f16(a, ld_frag(sT[cur] + 512 + lane * 16), acc1);
    __syncthreads();
    cur ^= 1;
  }
  const int half = lane >> 4, nl = lane & 15;
  v8f accs[2] = {acc0, acc1};
#pragma unroll
  for (int nt = 0; nt < 2; ++nt) {
#pragma unroll
    for (int r = 0; r < 8; ++r) {
      const int c = c_base + wv * 16 + r + 8 * half;
      out[((size_t)b * nC + c) * nN + n_base + nt * 16 + nl] = (_Float16)accs[nt][r];
    }
  }
}

// ---------- Kernel 4: msw 1x1 conv (K=2560) + bn + sigmoid -> weights[b,4,n] ----------
__global__ __launch_bounds__(256) void msw_kernel(
    const _Float16* __restrict__ outs, const float* __restrict__ mswW,
    const float* __restrict__ g, const float* __restrict__ bb,
    const float* __restrict__ m, const float* __restrict__ v,
    float* __restrict__ wgt) {
  const int tid = blockIdx.x * 256 + threadIdx.x;    // 0..32767
  const int n = tid & 255;
  const int o = (tid >> 8) & 3;
  const int b = tid >> 10;
  float s = 0.f;
  for (int hd = 0; hd < 5; ++hd) {
    const _Float16* ob = outs + ((size_t)hd * nB + b) * nC * nN + n;
    const float* wr = mswW + (size_t)o * (5 * nC) + hd * nC;
    for (int c = 0; c < nC; ++c) s += wr[c] * (float)ob[(size_t)c * nN];
  }
  const float tt = (s - m[o]) * (g[o] * rsqrtf(v[o] + 1e-5f)) + bb[o];
  wgt[tid] = 1.0f / (1.0f + __expf(-tt));
}

// ---------- Kernel 5: weighted combine -> zero-padded f16 image [B][C][18][18] ----------
__global__ __launch_bounds__(256) void combine_kernel(const _Float16* __restrict__ outs,
                                                      const float* __restrict__ wgt,
                                                      _Float16* __restrict__ xmsp) {
  const size_t tid = (size_t)blockIdx.x * 256 + threadIdx.x;  // B*C*18*18
  const int pw = (int)(tid % 18);
  size_t r = tid / 18;
  const int ph = (int)(r % 18); r /= 18;
  const int c = (int)(r % nC);
  const int b = (int)(r / nC);
  _Float16 val = (_Float16)0.f;
  if (ph >= 1 && ph <= 16 && pw >= 1 && pw <= 16) {
    const int n = (ph - 1) * 16 + (pw - 1);
    const size_t base = ((size_t)b * nC + c) * nN + n;
    float s = (float)outs[4 * BCN + base];
#pragma unroll
    for (int hd = 0; hd < 4; ++hd)
      s += wgt[(b * 4 + hd) * 256 + n] * (float)outs[(size_t)hd * BCN + base];
    val = (_Float16)s;
  }
  xmsp[tid] = val;
}

// ---------- Kernel 6: conv weight reorg  fus_W[o][ci][kh][kw] -> f16 [tap][o][ci] ----------
__global__ __launch_bounds__(256) void wreorg_kernel(const float* __restrict__ fusW,
                                                     _Float16* __restrict__ convW) {
  const size_t tid = (size_t)blockIdx.x * 256 + threadIdx.x;  // 9*C*C
  const int ci = (int)(tid % nC);
  size_t r = tid / nC;
  const int o = (int)(r % nC);
  const int tap = (int)(r / nC);
  convW[tid] = (_Float16)fusW[((size_t)o * nC + ci) * 9 + tap];
}

// ---------- Kernel 7: 3x3 conv as 9 GEMM taps + bn+relu + residual + bn+relu ----------
// grid (8 hrow-pairs, 4 c0-tiles, 32 b), block 256; 144 pipelined k-steps
__global__ __launch_bounds__(256) void conv_kernel(
    const _Float16* __restrict__ xmsp, const _Float16* __restrict__ convW,
    const float* __restrict__ xres,
    const float* __restrict__ fg, const float* __restrict__ fb,
    const float* __restrict__ fm, const float* __restrict__ fv,
    const float* __restrict__ tg, const float* __restrict__ tb,
    const float* __restrict__ tm, const float* __restrict__ tv,
    float* __restrict__ out) {
  __shared__ __align__(16) _Float16 sA[2][8 * 512];
  __shared__ __align__(16) _Float16 sB[2][2 * 512];
  const int hrow0 = blockIdx.x * 2;
  const int c_base = blockIdx.y * 128;
  const int b = blockIdx.z;
  const int t = threadIdx.x;
  const int wv = t >> 5, lane = t & 31;

  auto stage = [&](int it, _Float16* dA, _Float16* dB) {
    const int tap = it >> 4;
    const int kc = (it & 15) * 32;
    const int th = tap / 3, tw = tap % 3;
    {
      const int s = t >> 5, l = t & 31;
      const int m = l & 15, kh8 = (l >> 4) * 8;
      const _Float16* rowp =
          convW + (size_t)tap * nC * nC + (size_t)(c_base + s * 16 + m) * nC + kc;
      v16h f;
#pragma unroll
      for (int i = 0; i < 8; ++i) f[i] = rowp[kh8 + i];
#pragma unroll
      for (int i = 0; i < 8; ++i) f[8 + i] = rowp[16 + kh8 + i];
      st_frag(dA + s * 512 + l * 16, f);
    }
    {
      const int l = t & 31, eh = t >> 5;
      const int n0 = l & 15, k = (l >> 4) * 8 + eh * 2 + (eh >> 2) * 8;
#pragma unroll
      for (int nt = 0; nt < 2; ++nt) {
        const _Float16* base =
            xmsp + (((size_t)b * nC + kc + k) * 18 + hrow0 + nt + th) * 18 + tw + n0;
        st_pair(dB + nt * 512 + l * 16 + eh * 2, base[0], base[18 * 18]);
      }
    }
  };

  stage(0, sA[0], sB[0]);
  __syncthreads();
  v8f acc0 = {}, acc1 = {};
  int cur = 0;
  for (int it = 0; it < 144; ++it) {
    if (it + 1 < 144) stage(it + 1, sA[cur ^ 1], sB[cur ^ 1]);
    v16h a = ld_frag(sA[cur] + wv * 512 + lane * 16);
    acc0 = wmma_f16(a, ld_frag(sB[cur] + lane * 16), acc0);
    acc1 = wmma_f16(a, ld_frag(sB[cur] + 512 + lane * 16), acc1);
    __syncthreads();
    cur ^= 1;
  }
  const int half = lane >> 4, wl = lane & 15;
  v8f accs[2] = {acc0, acc1};
#pragma unroll
  for (int nt = 0; nt < 2; ++nt) {
#pragma unroll
    for (int r = 0; r < 8; ++r) {
      const int c = c_base + wv * 16 + r + 8 * half;
      const int n = (hrow0 + nt) * 16 + wl;
      float s = accs[nt][r];
      s = (s - fm[c]) * (fg[c] * rsqrtf(fv[c] + 1e-5f)) + fb[c];
      s = fmaxf(s, 0.f);
      s += xres[((size_t)b * nC + c) * nN + n];
      s = (s - tm[c]) * (tg[c] * rsqrtf(tv[c] + 1e-5f)) + tb[c];
      out[((size_t)b * nC + c) * nN + n] = fmaxf(s, 0.f);
    }
  }
}

// ---------- host launch ----------
extern "C" void kernel_launch(void* const* d_in, const int* in_sizes, int n_in,
                              void* d_out, int out_size, void* d_ws, size_t ws_size,
                              hipStream_t stream) {
  (void)in_sizes; (void)n_in; (void)out_size; (void)ws_size;
  const float* xin[5] = {(const float*)d_in[0], (const float*)d_in[1],
                         (const float*)d_in[2], (const float*)d_in[3],
                         (const float*)d_in[4]};
  const float* qW = (const float*)d_in[5];
  const float* qB = (const float*)d_in[6];
  const float* kW = (const float*)d_in[7];
  const float* kB = (const float*)d_in[8];
  const float* vW = (const float*)d_in[9];
  const float* vB = (const float*)d_in[10];
  const float* relH = (const float*)d_in[11];
  const float* relW = (const float*)d_in[12];
  const float* mswW = (const float*)d_in[13];
  const float* mswg = (const float*)d_in[14];
  const float* mswb = (const float*)d_in[15];
  const float* mswm = (const float*)d_in[16];
  const float* mswv = (const float*)d_in[17];
  const float* fusW = (const float*)d_in[18];
  const float* fg = (const float*)d_in[19];
  const float* fb = (const float*)d_in[20];
  const float* fm = (const float*)d_in[21];
  const float* fv = (const float*)d_in[22];
  const float* tg = (const float*)d_in[23];
  const float* tb = (const float*)d_in[24];
  const float* tm = (const float*)d_in[25];
  const float* tv = (const float*)d_in[26];

  // workspace layout (bytes)
  char* ws = (char*)d_ws;
  _Float16* q16    = (_Float16*)(ws + 0);                     //  8,388,608
  _Float16* k16    = (_Float16*)(ws + 8388608);               //  8,388,608
  _Float16* v16    = (_Float16*)(ws + 16777216);              //  8,388,608
  _Float16* attn16 = (_Float16*)(ws + 25165824);              //  4,194,304
  _Float16* outs16 = (_Float16*)(ws + 29360128);              // 41,943,040
  float*    wgt    = (float*)   (ws + 71303168);              //    131,072
  _Float16* xmsp   = (_Float16*)(ws + 71434240);              // 10,616,832
  _Float16* convW16= (_Float16*)(ws + 82051072);              //  4,718,592
                                                              // total ~86.8 MB

  const dim3 gGemm(8, 4, 32);
  for (int hd = 0; hd < 5; ++hd) {
    const size_t wOff = (size_t)hd * nC * nC;
    qkv_kernel<<<gGemm, 256, 0, stream>>>(xin[hd], qW + wOff, qB + hd * nC,
                                          nullptr, nullptr, 0, q16);
    qkv_kernel<<<gGemm, 256, 0, stream>>>(xin[hd], kW + wOff, kB + hd * nC,
                                          relH + (size_t)hd * nC * nH,
                                          relW + (size_t)hd * nC * nW, 1, k16);
    qkv_kernel<<<gGemm, 256, 0, stream>>>(xin[hd], vW + wOff, vB + hd * nC,
                                          nullptr, nullptr, 0, v16);
    attn_kernel<<<dim3(16, 32), 512, 0, stream>>>(q16, k16, attn16);
    av_kernel<<<gGemm, 256, 0, stream>>>(v16, attn16, outs16 + (size_t)hd * BCN);
  }
  wreorg_kernel<<<9216, 256, 0, stream>>>(fusW, convW16);
  msw_kernel<<<128, 256, 0, stream>>>(outs16, mswW, mswg, mswb, mswm, mswv, wgt);
  combine_kernel<<<20736, 256, 0, stream>>>(outs16, wgt, xmsp);
  conv_kernel<<<dim3(8, 4, 32), 256, 0, stream>>>(xmsp, convW16, xin[4],
                                                  fg, fb, fm, fv, tg, tb, tm, tv,
                                                  (float*)d_out);
}
